// model_60387240181960
// MI455X (gfx1250) — matrix operand order repaired and checked by reading, not compiled, not run on hardware
//
#include <hip/hip_runtime.h>

// ---- problem constants (match reference) ----
#define BATCH 16384
#define DIM   512
#define NOUT  50
#define BPB   8            // batch elements per block
#define ROWS  64           // 8*7 = 56 real rows + 8 pad

typedef __attribute__((ext_vector_type(16))) _Float16 v16h;
typedef __attribute__((ext_vector_type(8)))  _Float16 v8h;
typedef __attribute__((ext_vector_type(2)))  _Float16 h2;
typedef __attribute__((ext_vector_type(8)))  float    v8f;

// LDS layout (bytes)
#define OFF_XSH   0                         // 64*512 f16 = 65536
#define OFF_TSH   65536                     // 64*512 f16 = 65536
#define OFF_REAC  131072                    // 8*512 f32  = 16384
#define OFF_S     147456                    // 8*120 f32  = 3840
#define OFF_MD    151296                    // 8*16*2 f32 = 1024
#define OFF_ATT   152320                    // 8*16 f32   = 512
#define SMEM_BYTES 152832

// ============================================================
// K1: Mt[n*512+k] = sum_e Wq[k,e] * Wk[n,e]  (= (Wq @ Wk^T)^T), f16 out.
// Transposed so WMMA B-fragments are contiguous 16B runs.
// ============================================================
__global__ __launch_bounds__(256) void make_mt(const float* __restrict__ Wq,
                                               const float* __restrict__ Wk,
                                               _Float16* __restrict__ Mt) {
    int tid = blockIdx.x * 256 + threadIdx.x;        // 0 .. 512*512-1
    int k = tid & (DIM - 1);
    int n = tid >> 9;
    const float4* __restrict__ a = (const float4*)(Wq + (size_t)k * DIM);
    const float4* __restrict__ b = (const float4*)(Wk + (size_t)n * DIM);
    float acc = 0.f;
#pragma unroll 4
    for (int i = 0; i < DIM / 4; ++i) {
        float4 x = a[i], y = b[i];
        acc = fmaf(x.x, y.x, acc);
        acc = fmaf(x.y, y.y, acc);
        acc = fmaf(x.z, y.z, acc);
        acc = fmaf(x.w, y.w, acc);
    }
    Mt[(size_t)n * DIM + k] = (_Float16)acc;
}

// ---------------- attention helpers ----------------
__device__ inline void expand_r(int k, int& a, int& b) {
    const int pa[6] = {0, 0, 0, 1, 1, 2};
    const int pb[6] = {1, 2, 3, 2, 3, 3};
    if (k < 4) { a = k; b = -1; } else { a = pa[k - 4]; b = pb[k - 4]; }
}
__device__ inline void expand_p(int k, int& a, int& b) {
    const int pa[3] = {0, 0, 1};
    const int pb[3] = {1, 2, 2};
    if (k < 3) { a = k; b = -1; } else { a = pa[k - 3]; b = pb[k - 3]; }
}

__device__ inline float dot_expand(const _Float16* Q, int qa, int qb,
                                   const _Float16* K, int ka, int kb) {
    float acc = 0.f;
    for (int d = 0; d < DIM; d += 2) {
        h2 q = *(const h2*)(Q + qa * DIM + d);
        if (qb >= 0) { h2 t = *(const h2*)(Q + qb * DIM + d); q = q + t; }
        h2 kv = *(const h2*)(K + ka * DIM + d);
        if (kb >= 0) { h2 t = *(const h2*)(K + kb * DIM + d); kv = kv + t; }
        acc = fmaf((float)q.x, (float)kv.x, acc);
        acc = fmaf((float)q.y, (float)kv.y, acc);
    }
    return acc;
}

// ============================================================
// Fused: stage X (f16, LDS) -> T = X@M via WMMA (T stays in LDS) ->
// cross-attention + softmax-mean + reaction + out-projection.
// Block: 128 threads (4 waves), 8 batch elements.
// ============================================================
__global__ __launch_bounds__(128) void fused_attn(
    const float* __restrict__ R, const float* __restrict__ P,
    const _Float16* __restrict__ Mt,
    const float* __restrict__ Wp, const float* __restrict__ bpv,
    float* __restrict__ out, float* __restrict__ att_r_out,
    float* __restrict__ att_p_out, float* __restrict__ reac_out) {
    extern __shared__ char smem[];
    _Float16* Xsh   = (_Float16*)(smem + OFF_XSH);   // [64][512] f16
    _Float16* Tsh   = (_Float16*)(smem + OFF_TSH);   // [64][512] f16
    float*    shReac= (float*)(smem + OFF_REAC);     // [8][512]
    float*    shS   = (float*)(smem + OFF_S);        // [8][120]
    float*    shMD  = (float*)(smem + OFF_MD);       // [8][16][2]
    float*    shAtt = (float*)(smem + OFF_ATT);      // [8][16]

    const int tid = threadIdx.x;
    const int w = tid >> 5, lane = tid & 31;
    const int bBase = blockIdx.x * BPB;

    // ---- Phase 0: stage 56 rows fp32 -> f16 LDS (rows: b_local*7 + mol) ----
    for (int idx = tid; idx < 56 * (DIM / 4); idx += 128) {
        const int lr = idx >> 7, c = idx & 127;          // 128 float4 chunks/row
        const int bl = lr / 7, mol = lr % 7;
        const int b = bBase + bl;
        const float* src = (mol < 4)
            ? (R + ((size_t)mol * BATCH + b) * DIM)
            : (P + ((size_t)(mol - 4) * BATCH + b) * DIM);
        float4 f = ((const float4*)src)[c];
        h2 lo = {(_Float16)f.x, (_Float16)f.y};
        h2 hi = {(_Float16)f.z, (_Float16)f.w};
        *(h2*)(Xsh + lr * DIM + c * 4)     = lo;
        *(h2*)(Xsh + lr * DIM + c * 4 + 2) = hi;
    }
    for (int idx = tid; idx < 8 * DIM / 2; idx += 128) { // zero pad rows 56..63
        h2 z = {(_Float16)0.f, (_Float16)0.f};
        *(h2*)(Xsh + 56 * DIM + idx * 2) = z;
    }
    __syncthreads();

    // ---- Phase 1: T = X @ M  (wave w owns rows [w*16, w*16+16)) ----
    const int m = lane & 15, kh = lane >> 4;
    const _Float16* arow = Xsh + (size_t)(w * 16 + m) * DIM;
    for (int ct = 0; ct < DIM / 64; ++ct) {
        const int colBase = ct * 64;
        v8f acc[4] = {};
        for (int kk = 0; kk < DIM; kk += 32) {
            // A fragment (16x32 f16): two contiguous 8-half runs per lane
            v8h alo = *(const v8h*)(arow + kk + kh * 8);
            v8h ahi = *(const v8h*)(arow + kk + 16 + kh * 8);
            v16h a;
#pragma unroll
            for (int i = 0; i < 8; ++i) { a[i] = alo[i]; a[8 + i] = ahi[i]; }
            // 4 B fragments (32x16) from L2-resident Mt
#pragma unroll
            for (int nt = 0; nt < 4; ++nt) {
                const _Float16* bp =
                    Mt + (size_t)(colBase + nt * 16 + m) * DIM + kk + kh * 8;
                v8h lo = *(const v8h*)bp;
                v8h hi = *(const v8h*)(bp + 16);
                v16h bfrag;
#pragma unroll
                for (int i = 0; i < 8; ++i) { bfrag[i] = lo[i]; bfrag[8 + i] = hi[i]; }
                acc[nt] = __builtin_amdgcn_wmma_f32_16x16x32_f16(
                    false, a, false, bfrag, (short)0, acc[nt], false, false);
            }
        }
        // spill D to LDS as f16 (row = w*16 + v + 8*kh, col = colBase+nt*16+m)
#pragma unroll
        for (int nt = 0; nt < 4; ++nt) {
            const int n = colBase + nt * 16 + m;
#pragma unroll
            for (int v = 0; v < 8; ++v)
                Tsh[(size_t)(w * 16 + v + 8 * kh) * DIM + n] = (_Float16)acc[nt][v];
        }
    }
    __syncthreads();

    // ---- Phase 2: attention, 2 batch elements per wave ----
    const float kscale = 0.04419417382415922f;  // 1/sqrt(512)
    for (int sub = 0; sub < 2; ++sub) {
        const int bl = w * 2 + sub;
        const int b = bBase + bl;
        const _Float16* Xr = Xsh + (size_t)(bl * 7) * DIM;      // 4 reactant rows
        const _Float16* Xp = Xr + 4 * DIM;                      // 3 product rows
        const _Float16* Tr = Tsh + (size_t)(bl * 7) * DIM;      // Rx@M base rows
        const _Float16* Tp = Tr + 4 * DIM;                      // Px@M base rows
        float* S   = shS   + bl * 120;
        float* MD  = shMD  + bl * 32;
        float* Att = shAtt + bl * 16;
        float* Rc  = shReac + bl * DIM;

        // 120 expanded dots
        for (int s = 0; s < 4; ++s) {
            int e = lane + 32 * s;
            if (e < 120) {
                int qa, qb, ka, kb; float v;
                if (e < 60) {              // scores_r[q<6, k<10] = TPx[q].Rx[k]
                    int q = e / 10, k = e % 10;
                    expand_p(q, qa, qb); expand_r(k, ka, kb);
                    v = dot_expand(Tp, qa, qb, Xr, ka, kb);
                } else {                   // scores_p[q<10, k<6] = TRx[q].Px[k]
                    int e2 = e - 60, q = e2 / 6, k = e2 % 6;
                    expand_r(q, qa, qb); expand_p(k, ka, kb);
                    v = dot_expand(Tr, qa, qb, Xp, ka, kb);
                }
                S[e] = v * kscale;
            }
        }
        __syncthreads();

        // per-query max / denom
        if (lane < 6) {
            float mx = -3.4e38f;
            for (int k = 0; k < 10; ++k) mx = fmaxf(mx, S[lane * 10 + k]);
            float sm = 0.f;
            for (int k = 0; k < 10; ++k) sm += __expf(S[lane * 10 + k] - mx);
            MD[lane * 2] = mx; MD[lane * 2 + 1] = sm;
        } else if (lane >= 16 && lane < 26) {
            int q = lane - 16;
            float mx = -3.4e38f;
            for (int k = 0; k < 6; ++k) mx = fmaxf(mx, S[60 + q * 6 + k]);
            float sm = 0.f;
            for (int k = 0; k < 6; ++k) sm += __expf(S[60 + q * 6 + k] - mx);
            MD[(6 + q) * 2] = mx; MD[(6 + q) * 2 + 1] = sm;
        }
        __syncthreads();

        // softmax over keys, mean over queries
        if (lane < 10) {
            float a = 0.f;
            for (int q = 0; q < 6; ++q)
                a += __expf(S[q * 10 + lane] - MD[q * 2]) / MD[q * 2 + 1];
            a *= (1.f / 6.f);
            Att[lane] = a;
            att_r_out[(size_t)b * 10 + lane] = a;
        }
        if (lane >= 16 && lane < 22) {
            int k = lane - 16;
            float a = 0.f;
            for (int q = 0; q < 10; ++q)
                a += __expf(S[60 + q * 6 + k] - MD[(6 + q) * 2]) / MD[(6 + q) * 2 + 1];
            a *= 0.1f;
            Att[10 + k] = a;
            att_p_out[(size_t)b * 6 + k] = a;
        }
        __syncthreads();

        // collapse expanded attention onto base rows, form reaction
        const float cr0 = Att[0] + Att[4] + Att[5] + Att[6];
        const float cr1 = Att[1] + Att[4] + Att[7] + Att[8];
        const float cr2 = Att[2] + Att[5] + Att[7] + Att[9];
        const float cr3 = Att[3] + Att[6] + Att[8] + Att[9];
        const float cp0 = Att[10] + Att[13] + Att[14];
        const float cp1 = Att[11] + Att[13] + Att[15];
        const float cp2 = Att[12] + Att[14] + Att[15];
        for (int i = 0; i < DIM / 32; ++i) {
            const int d = lane + 32 * i;
            float r = cr0 * (float)Xr[0 * DIM + d] + cr1 * (float)Xr[1 * DIM + d] +
                      cr2 * (float)Xr[2 * DIM + d] + cr3 * (float)Xr[3 * DIM + d] -
                      (cp0 * (float)Xp[0 * DIM + d] + cp1 * (float)Xp[1 * DIM + d] +
                       cp2 * (float)Xp[2 * DIM + d]);
            Rc[d] = r;
            reac_out[(size_t)b * DIM + d] = r;
        }
        __syncthreads();

        // out = reaction @ W_pred + b_pred
        for (int o = lane; o < NOUT; o += 32) {
            float acc = bpv[o];
            for (int d = 0; d < DIM; ++d)
                acc = fmaf(Rc[d], Wp[(size_t)d * NOUT + o], acc);
            out[(size_t)b * NOUT + o] = acc;
        }
    }
}

// ============================================================
extern "C" void kernel_launch(void* const* d_in, const int* in_sizes, int n_in,
                              void* d_out, int out_size, void* d_ws, size_t ws_size,
                              hipStream_t stream) {
    const float* R  = (const float*)d_in[0];  // [4, B, 512]
    const float* P  = (const float*)d_in[1];  // [3, B, 512]
    const float* Wq = (const float*)d_in[2];  // [512, 512]
    const float* Wk = (const float*)d_in[3];  // [512, 512]
    const float* Wp = (const float*)d_in[4];  // [512, 50]
    const float* bp = (const float*)d_in[5];  // [50]

    _Float16* Mt = (_Float16*)d_ws;           // only 512 KB of scratch needed

    float* out  = (float*)d_out;                    // [B, 50]
    float* attr = out  + (size_t)BATCH * NOUT;      // [B, 10]
    float* attp = attr + (size_t)BATCH * 10;        // [B, 6]
    float* reac = attp + (size_t)BATCH * 6;         // [B, 512]

    (void)hipFuncSetAttribute((const void*)fused_attn,
                              hipFuncAttributeMaxDynamicSharedMemorySize,
                              SMEM_BYTES);

    make_mt<<<(DIM * DIM) / 256, 256, 0, stream>>>(Wq, Wk, Mt);
    fused_attn<<<BATCH / BPB, 128, SMEM_BYTES, stream>>>(R, P, Mt, Wp, bp,
                                                         out, attr, attp, reac);
}
